// QILMTransformerBlock_9019431322191
// MI455X (gfx1250) — matrix operand order repaired and checked by reading, not compiled, not run on hardware
//
#include <hip/hip_runtime.h>
#include <hip/hip_bf16.h>
#include <math.h>

// ---------------------------------------------------------------------------
// QILM complex transformer block for MI455X (gfx1250, wave32, WMMA).
// All GEMMs run through v_wmma_f32_16x16x32_bf16 (f32 accumulate).
// Complex linears are folded: A=[xr;xi] (2M x K) x W=[wr;wi] (2N x K)^T gives
// all four real GEMMs in one launch; combine kernels form yr/yi (+bias).
// GEMM uses double-buffered LDS tiles filled with GLOBAL_LOAD_ASYNC_TO_LDS
// (ASYNCcnt) -- builtins confirmed present on this toolchain.
// Workspace plan (~740 MB; assumes harness ws is large).
// ---------------------------------------------------------------------------

typedef __attribute__((ext_vector_type(16))) __bf16 v16bf;
typedef __attribute__((ext_vector_type(8)))  float  v8f;

constexpr int Bc = 2, Tc = 2048, Dc = 1024, Hc = 16, FFc = 4096;
constexpr int Mrows = Bc * Tc;       // 4096 token rows
constexpr int BH    = Bc * Hc;       // 32 attention batches
constexpr int CH    = 8;             // head-batches per attention chunk

#define BLK_M 128
#define BLK_N 128
#define BLK_K 32

#if __has_builtin(__builtin_amdgcn_global_load_async_to_lds_b128) && \
    __has_builtin(__builtin_amdgcn_s_wait_asynccnt)
#define USE_ASYNC_LDS 1
#else
#define USE_ASYNC_LDS 0
#endif

#if USE_ASYNC_LDS
typedef int async_v4i __attribute__((vector_size(4 * sizeof(int))));
typedef async_v4i __attribute__((address_space(1)))* async_gp;  // global
typedef async_v4i __attribute__((address_space(3)))* async_lp;  // LDS
#endif

// 16-byte global -> LDS copy; async (no VGPR round-trip, ASYNCcnt) if available
__device__ __forceinline__ void copy16_g2l(const __bf16* __restrict__ g,
                                           __bf16* __restrict__ l) {
#if USE_ASYNC_LDS
  __builtin_amdgcn_global_load_async_to_lds_b128(
      (async_gp)(void*)g, (async_lp)(void*)l, 0, 0);
#else
  *(uint4*)l = *(const uint4*)g;
#endif
}

__device__ __forceinline__ void wait_g2l() {
#if USE_ASYNC_LDS
  __builtin_amdgcn_s_wait_asynccnt(0);
#endif
}

// ---------------------------------------------------------------------------
// Tiled bf16 GEMM:  C[m,n] = sum_k A[m,k] * B[n,k]   (A: MxK, B: NxK, both
// row-major, i.e. computes A x B^T).  Batched via blockIdx.z with element
// strides sA/sB/sC.  OUT_BF16 selects fp32 or bf16 output.
// 256 threads = 8 wave32s; block tile 128x128x32; wave tile 32x64 (2x4 WMMA).
// Double-buffered LDS: tile k+1 streams in while tile k is consumed.
// ---------------------------------------------------------------------------
template <bool OUT_BF16>
__global__ __launch_bounds__(256) void gemm_bf16_kernel(
    const __bf16* __restrict__ A, const __bf16* __restrict__ B,
    void* __restrict__ Cout, int N, int K,
    long long sA, long long sB, long long sC) {
  __shared__ __align__(16) __bf16 As[2][BLK_M * BLK_K];
  __shared__ __align__(16) __bf16 Bs[2][BLK_N * BLK_K];

  const int tid  = threadIdx.x;
  const int lane = tid & 31;
  const int wave = tid >> 5;
  const int wm = (wave >> 1) * 32;   // wave row offset inside block tile (4 waves in M)
  const int wn = (wave & 1) * 64;    // wave col offset inside block tile (2 waves in N)

  const int m0 = blockIdx.y * BLK_M;
  const int n0 = blockIdx.x * BLK_N;
  const size_t batch = blockIdx.z;

  const __bf16* Ab = A + batch * (size_t)sA;
  const __bf16* Bb = B + batch * (size_t)sB;

  v8f acc[2][4] = {};

  const int lr = tid >> 2;           // 0..63 : tile row for LDS fill
  const int lc = (tid & 3) * 8;      // 0,8,16,24 : K offset (8 bf16 = 16B)
  const int nk = K / BLK_K;

  // issue global->LDS copies for K-tile kt into LDS buffer `buf`
  auto issue_tile = [&](int kt, int buf) {
    const int k0 = kt * BLK_K;
    const __bf16* gA0 = Ab + (size_t)(m0 + lr) * K + k0 + lc;
    const __bf16* gA1 = Ab + (size_t)(m0 + lr + 64) * K + k0 + lc;
    const __bf16* gB0 = Bb + (size_t)(n0 + lr) * K + k0 + lc;
    const __bf16* gB1 = Bb + (size_t)(n0 + lr + 64) * K + k0 + lc;
    copy16_g2l(gA0, &As[buf][lr * BLK_K + lc]);
    copy16_g2l(gA1, &As[buf][(lr + 64) * BLK_K + lc]);
    copy16_g2l(gB0, &Bs[buf][lr * BLK_K + lc]);
    copy16_g2l(gB1, &Bs[buf][(lr + 64) * BLK_K + lc]);
#if !USE_ASYNC_LDS
    if (k0 + BLK_K < K) {            // pull next K-tile toward near caches
      __builtin_prefetch(gA0 + BLK_K, 0, 3);
      __builtin_prefetch(gB0 + BLK_K, 0, 3);
    }
#endif
  };

  issue_tile(0, 0);

  for (int kt = 0; kt < nk; ++kt) {
    const int cur = kt & 1;
    // Per-wave: our async copies for buffer `cur` have landed.
    wait_g2l();
    // All waves: buffer `cur` fully written AND everyone is done reading the
    // buffer we are about to refill (they passed their previous compute).
    __syncthreads();
    if (kt + 1 < nk) issue_tile(kt + 1, cur ^ 1);

    // A fragment: 16-bit A 16x32 layout (ISA 7.12.2): lane&15 = row,
    // lane>>4 selects K sub-banks; VGPR g holds K = (g<4?0:16)+hi*8+(g&3)*2.
    v16bf af[2];
#pragma unroll
    for (int mt = 0; mt < 2; ++mt) {
      const __bf16* p = &As[cur][(wm + mt * 16 + (lane & 15)) * BLK_K];
      const int hi = (lane >> 4) * 8;
#pragma unroll
      for (int g = 0; g < 8; ++g) {
        const int kk = ((g < 4) ? 0 : 16) + hi + (g & 3) * 2;
        af[mt][2 * g]     = p[kk];
        af[mt][2 * g + 1] = p[kk + 1];
      }
    }
    // B fragment: lane&15 = output column, same K order as A:
    // element g holds K = (g<8?0:16) + (lane>>4)*8 + (g&7).
    v16bf bfv[4];
#pragma unroll
    for (int nt = 0; nt < 4; ++nt) {
      const __bf16* p = &Bs[cur][(wn + nt * 16 + (lane & 15)) * BLK_K + (lane >> 4) * 8];
#pragma unroll
      for (int g = 0; g < 16; ++g) bfv[nt][g] = p[(g < 8) ? g : g + 8];
    }
#pragma unroll
    for (int mt = 0; mt < 2; ++mt)
#pragma unroll
      for (int nt = 0; nt < 4; ++nt)
        acc[mt][nt] = __builtin_amdgcn_wmma_f32_16x16x32_bf16(
            false, af[mt], false, bfv[nt], (short)0, acc[mt][nt], false, false);
  }

  // C/D layout: lane&15 = N, VGPR g = M row g + (lane>>4)*8.
#pragma unroll
  for (int mt = 0; mt < 2; ++mt) {
#pragma unroll
    for (int nt = 0; nt < 4; ++nt) {
      const int n = n0 + wn + nt * 16 + (lane & 15);
      const int mbase = m0 + wm + mt * 16 + (lane >> 4) * 8;
#pragma unroll
      for (int g = 0; g < 8; ++g) {
        const size_t idx = batch * (size_t)sC + (size_t)(mbase + g) * N + n;
        if constexpr (OUT_BF16) ((__bf16*)Cout)[idx] = (__bf16)acc[mt][nt][g];
        else                    ((float*)Cout)[idx]  = acc[mt][nt][g];
      }
    }
  }
}

// ---------------------------------------------------------------------------
// Elementwise / reduction kernels
// ---------------------------------------------------------------------------
__device__ inline float block_reduce(float v, float* sred, int tid, bool is_max) {
  sred[tid] = v;
  __syncthreads();
  for (int s = 128; s > 0; s >>= 1) {
    if (tid < s) {
      const float o = sred[tid + s];
      sred[tid] = is_max ? fmaxf(sred[tid], o) : (sred[tid] + o);
    }
    __syncthreads();
  }
  const float r = sred[0];
  __syncthreads();
  return r;
}

// stack two fp32 arrays into one bf16 buffer: dst[0:n]=a, dst[n:2n]=b
__global__ void pack2_bf16_kernel(const float* __restrict__ a,
                                  const float* __restrict__ b,
                                  __bf16* __restrict__ dst, size_t n) {
  const size_t i = (size_t)blockIdx.x * 256 + threadIdx.x;
  if (i < n) {
    dst[i]     = (__bf16)a[i];
    dst[n + i] = (__bf16)b[i];
  }
}

// rotary on first D/2 features; emits xr fp32 (for residual) + [xr;xi] bf16 stack
__global__ __launch_bounds__(256) void rotary_kernel(
    const float* __restrict__ xre, const float* __restrict__ xim,
    float* __restrict__ xrf, __bf16* __restrict__ xstack) {
  const int row = blockIdx.x;                 // 0..Mrows-1
  const int t   = row % Tc;
  const int tid = threadIdx.x;
  const float LOG1E4 = 9.210340371976184f;    // log(10000)
#pragma unroll
  for (int k = 0; k < Dc / 256; ++k) {
    const int j = tid + k * 256;
    const float vr = xre[(size_t)row * Dc + j];
    const float vi = xim[(size_t)row * Dc + j];
    float rr = vr, ri = vi;
    if (j < Dc / 2) {
      const float fr = __expf(-LOG1E4 * (float)j / (float)(Dc / 2));
      const float ang = (float)t * fr;
      const float c = __cosf(ang), s = __sinf(ang);
      rr = vr * c - vi * s;
      ri = vr * s + vi * c;
    }
    xrf[(size_t)row * Dc + j] = rr;
    xstack[(size_t)row * Dc + j]            = (__bf16)rr;
    xstack[(size_t)(Mrows + row) * Dc + j]  = (__bf16)ri;
  }
}

// combine stacked-GEMM output (2M x 2D) into per-head packed q/k (b,h,t,128)
// or transposed v (b,h,128,t).  yr=G1-G2+br-bi, yi=G3+G4+br+bi.
__global__ void combine_pack_qk_kernel(const float* __restrict__ C2,
                                       const float* __restrict__ br,
                                       const float* __restrict__ bi,
                                       __bf16* __restrict__ out, int transposed) {
  const size_t i = (size_t)blockIdx.x * 256 + threadIdx.x;
  if (i >= (size_t)Mrows * Dc) return;
  const int m = (int)(i / Dc), n = (int)(i % Dc);
  const float g1 = C2[(size_t)m * (2 * Dc) + n];
  const float g4 = C2[(size_t)m * (2 * Dc) + Dc + n];
  const float g3 = C2[(size_t)(Mrows + m) * (2 * Dc) + n];
  const float g2 = C2[(size_t)(Mrows + m) * (2 * Dc) + Dc + n];
  const float yr = g1 - g2 + br[n] - bi[n];
  const float yi = g3 + g4 + br[n] + bi[n];
  const int b = m / Tc, t = m % Tc, h = n >> 6, dd = n & 63;
  if (!transposed) {
    const size_t base = ((size_t)(b * Hc + h) * Tc + t) * 128;
    out[base + dd]      = (__bf16)yr;
    out[base + 64 + dd] = (__bf16)yi;
  } else {
    const size_t base = (size_t)(b * Hc + h) * 128 * Tc;
    out[base + (size_t)dd * Tc + t]        = (__bf16)yr;
    out[base + (size_t)(64 + dd) * Tc + t] = (__bf16)yi;
  }
}

// generic combine for o-proj / f2: fp32 yr/yi (+ optional bf16 [yr;yi] stack)
__global__ void combine_linear_kernel(const float* __restrict__ C2,
                                      const float* __restrict__ br,
                                      const float* __restrict__ bi,
                                      float* __restrict__ yrf, float* __restrict__ yif,
                                      __bf16* __restrict__ stack, int N) {
  const size_t i = (size_t)blockIdx.x * 256 + threadIdx.x;
  if (i >= (size_t)Mrows * N) return;
  const int m = (int)(i / N), n = (int)(i % N);
  const float g1 = C2[(size_t)m * (2 * N) + n];
  const float g4 = C2[(size_t)m * (2 * N) + N + n];
  const float g3 = C2[(size_t)(Mrows + m) * (2 * N) + n];
  const float g2 = C2[(size_t)(Mrows + m) * (2 * N) + N + n];
  const float yr = g1 - g2 + br[n] - bi[n];
  const float yi = g3 + g4 + br[n] + bi[n];
  if (yrf) yrf[(size_t)m * N + n] = yr;
  if (yif) yif[(size_t)m * N + n] = yi;
  if (stack) {
    stack[(size_t)m * N + n]             = (__bf16)yr;
    stack[(size_t)(Mrows + m) * N + n]   = (__bf16)yi;
  }
}

// f1 combine with exact gelu on real part, emits bf16 [gelu(mr); mi] stack
__global__ void combine_ffn_gelu_kernel(const float* __restrict__ C2,
                                        const float* __restrict__ br,
                                        const float* __restrict__ bi,
                                        __bf16* __restrict__ stack) {
  const size_t i = (size_t)blockIdx.x * 256 + threadIdx.x;
  if (i >= (size_t)Mrows * FFc) return;
  const int m = (int)(i / FFc), n = (int)(i % FFc);
  const float g1 = C2[(size_t)m * (2 * FFc) + n];
  const float g4 = C2[(size_t)m * (2 * FFc) + FFc + n];
  const float g3 = C2[(size_t)(Mrows + m) * (2 * FFc) + n];
  const float g2 = C2[(size_t)(Mrows + m) * (2 * FFc) + FFc + n];
  const float yr = g1 - g2 + br[n] - bi[n];
  const float yi = g3 + g4 + br[n] + bi[n];
  const float gel = 0.5f * yr * (1.0f + erff(yr * 0.7071067811865476f));
  stack[(size_t)m * FFc + n]           = (__bf16)gel;
  stack[(size_t)(Mrows + m) * FFc + n] = (__bf16)yi;
}

// row softmax over T=2048 (8 elems/thread), bf16 in -> bf16 out, scale=1/sqrt(d)
__global__ __launch_bounds__(256) void softmax_bf16_kernel(
    const __bf16* __restrict__ scores, __bf16* __restrict__ probs, float scale) {
  __shared__ float sred[256];
  const int tid = threadIdx.x;
  const size_t row = blockIdx.x;
  const __bf16* in = scores + row * (size_t)Tc;
  __bf16* out = probs + row * (size_t)Tc;
  float v[8];
  float mx = -3.0e38f;
#pragma unroll
  for (int k = 0; k < 8; ++k) {
    v[k] = scale * (float)in[tid + k * 256];
    mx = fmaxf(mx, v[k]);
  }
  mx = block_reduce(mx, sred, tid, true);
  float sum = 0.f;
#pragma unroll
  for (int k = 0; k < 8; ++k) { v[k] = __expf(v[k] - mx); sum += v[k]; }
  sum = block_reduce(sum, sred, tid, false);
  const float inv = 1.0f / sum;
#pragma unroll
  for (int k = 0; k < 8; ++k) out[tid + k * 256] = (__bf16)(v[k] * inv);
}

// unpack attention output (b,h,t,128) fp32 -> bf16 stack [ar;ai] (2M x D)
__global__ void unpack_attn_kernel(const float* __restrict__ apack,
                                   __bf16* __restrict__ arstack) {
  const size_t i = (size_t)blockIdx.x * 256 + threadIdx.x;
  if (i >= (size_t)Mrows * Dc) return;
  const int m = (int)(i / Dc), n = (int)(i % Dc);
  const int b = m / Tc, t = m % Tc, h = n >> 6, dd = n & 63;
  const size_t base = ((size_t)(b * Hc + h) * Tc + t) * 128;
  arstack[(size_t)m * Dc + n]           = (__bf16)apack[base + dd];
  arstack[(size_t)(Mrows + m) * Dc + n] = (__bf16)apack[base + 64 + dd];
}

// out = layernorm(a+b2)*g + be over D=1024 (4 elems/thread)
__global__ __launch_bounds__(256) void add_ln_kernel(
    const float* __restrict__ a, const float* __restrict__ b2,
    const float* __restrict__ g, const float* __restrict__ be,
    float* __restrict__ out) {
  __shared__ float sred[256];
  const int row = blockIdx.x, tid = threadIdx.x;
  float x[4];
  float s = 0.f;
#pragma unroll
  for (int k = 0; k < 4; ++k) {
    const int j = tid + k * 256;
    x[k] = a[(size_t)row * Dc + j] + b2[(size_t)row * Dc + j];
    s += x[k];
  }
  const float mu = block_reduce(s, sred, tid, false) * (1.0f / Dc);
  float vs = 0.f;
#pragma unroll
  for (int k = 0; k < 4; ++k) { const float d = x[k] - mu; vs += d * d; }
  const float var = block_reduce(vs, sred, tid, false) * (1.0f / Dc);
  const float inv = rsqrtf(var + 1e-5f);
#pragma unroll
  for (int k = 0; k < 4; ++k) {
    const int j = tid + k * 256;
    out[(size_t)row * Dc + j] = (x[k] - mu) * inv * g[j] + be[j];
  }
}

// final: out[0] = LN(res1+fr)*g+be ; out[1] = ai2 + fi
__global__ __launch_bounds__(256) void final_kernel(
    const float* __restrict__ res1, const float* __restrict__ fr,
    const float* __restrict__ ai2, const float* __restrict__ fi,
    const float* __restrict__ g, const float* __restrict__ be,
    float* __restrict__ out) {
  __shared__ float sred[256];
  const int row = blockIdx.x, tid = threadIdx.x;
  float x[4];
  float s = 0.f;
#pragma unroll
  for (int k = 0; k < 4; ++k) {
    const int j = tid + k * 256;
    x[k] = res1[(size_t)row * Dc + j] + fr[(size_t)row * Dc + j];
    s += x[k];
  }
  const float mu = block_reduce(s, sred, tid, false) * (1.0f / Dc);
  float vs = 0.f;
#pragma unroll
  for (int k = 0; k < 4; ++k) { const float d = x[k] - mu; vs += d * d; }
  const float var = block_reduce(vs, sred, tid, false) * (1.0f / Dc);
  const float inv = rsqrtf(var + 1e-5f);
#pragma unroll
  for (int k = 0; k < 4; ++k) {
    const int j = tid + k * 256;
    const size_t idx = (size_t)row * Dc + j;
    out[idx] = (x[k] - mu) * inv * g[j] + be[j];
    out[(size_t)Mrows * Dc + idx] = ai2[idx] + fi[idx];
  }
}

// ---------------------------------------------------------------------------
// Orchestration
// ---------------------------------------------------------------------------
extern "C" void kernel_launch(void* const* d_in, const int* in_sizes, int n_in,
                              void* d_out, int out_size, void* d_ws, size_t ws_size,
                              hipStream_t stream) {
  (void)in_sizes; (void)n_in; (void)out_size; (void)ws_size;
  const float* x_real = (const float*)d_in[0];
  const float* x_imag = (const float*)d_in[1];
  const float* qwr = (const float*)d_in[2];  const float* qwi = (const float*)d_in[3];
  const float* qbr = (const float*)d_in[4];  const float* qbi = (const float*)d_in[5];
  const float* kwr = (const float*)d_in[6];  const float* kwi = (const float*)d_in[7];
  const float* kbr = (const float*)d_in[8];  const float* kbi = (const float*)d_in[9];
  const float* vwr = (const float*)d_in[10]; const float* vwi = (const float*)d_in[11];
  const float* vbr = (const float*)d_in[12]; const float* vbi = (const float*)d_in[13];
  const float* owr = (const float*)d_in[14]; const float* owi = (const float*)d_in[15];
  const float* obr = (const float*)d_in[16]; const float* obi = (const float*)d_in[17];
  const float* f1wr = (const float*)d_in[18]; const float* f1wi = (const float*)d_in[19];
  const float* f1br = (const float*)d_in[20]; const float* f1bi = (const float*)d_in[21];
  const float* f2wr = (const float*)d_in[22]; const float* f2wi = (const float*)d_in[23];
  const float* f2br = (const float*)d_in[24]; const float* f2bi = (const float*)d_in[25];
  const float* n1g = (const float*)d_in[26]; const float* n1b = (const float*)d_in[27];
  const float* n2g = (const float*)d_in[28]; const float* n2b = (const float*)d_in[29];

  char* wsb = (char*)d_ws;
  size_t off = 0;
  auto alloc = [&](size_t bytes) -> char* {
    char* p = wsb + off;
    off += (bytes + 255) & ~(size_t)255;
    return p;
  };

  __bf16* wqs  = (__bf16*)alloc((size_t)2 * Dc * Dc * 2);
  __bf16* wks  = (__bf16*)alloc((size_t)2 * Dc * Dc * 2);
  __bf16* wvs  = (__bf16*)alloc((size_t)2 * Dc * Dc * 2);
  __bf16* wos  = (__bf16*)alloc((size_t)2 * Dc * Dc * 2);
  __bf16* wf1s = (__bf16*)alloc((size_t)2 * FFc * Dc * 2);
  __bf16* wf2s = (__bf16*)alloc((size_t)2 * Dc * FFc * 2);
  __bf16* xstack  = (__bf16*)alloc((size_t)2 * Mrows * Dc * 2);
  float*  xrf     = (float*)alloc((size_t)Mrows * Dc * 4);
  __bf16* qpack   = (__bf16*)alloc((size_t)BH * Tc * 128 * 2);
  __bf16* kpack   = (__bf16*)alloc((size_t)BH * Tc * 128 * 2);
  __bf16* vpackT  = (__bf16*)alloc((size_t)BH * Tc * 128 * 2);
  float*  apack   = (float*)alloc((size_t)BH * Tc * 128 * 4);
  __bf16* scoresb = (__bf16*)alloc((size_t)CH * Tc * Tc * 2);
  __bf16* probsb  = (__bf16*)alloc((size_t)CH * Tc * Tc * 2);
  __bf16* arstack = (__bf16*)alloc((size_t)2 * Mrows * Dc * 2);
  float*  ar2f    = (float*)alloc((size_t)Mrows * Dc * 4);
  float*  ai2f    = (float*)alloc((size_t)Mrows * Dc * 4);
  __bf16* arstack2 = (__bf16*)alloc((size_t)2 * Mrows * Dc * 2);
  float*  res1    = (float*)alloc((size_t)Mrows * Dc * 4);
  __bf16* mstack  = (__bf16*)alloc((size_t)2 * Mrows * FFc * 2);
  float*  frf     = (float*)alloc((size_t)Mrows * Dc * 4);
  float*  fif     = (float*)alloc((size_t)Mrows * Dc * 4);
  float*  gemmC   = (float*)alloc((size_t)(2 * Mrows) * (2 * FFc) * 4);  // 256 MB, reused

  // --- weight casts into stacked bf16 [wr;wi] ---
  {
    const size_t nqk = (size_t)Dc * Dc;
    const int bqk = (int)((nqk + 255) / 256);
    pack2_bf16_kernel<<<bqk, 256, 0, stream>>>(qwr, qwi, wqs, nqk);
    pack2_bf16_kernel<<<bqk, 256, 0, stream>>>(kwr, kwi, wks, nqk);
    pack2_bf16_kernel<<<bqk, 256, 0, stream>>>(vwr, vwi, wvs, nqk);
    pack2_bf16_kernel<<<bqk, 256, 0, stream>>>(owr, owi, wos, nqk);
    const size_t nff = (size_t)FFc * Dc;
    const int bff = (int)((nff + 255) / 256);
    pack2_bf16_kernel<<<bff, 256, 0, stream>>>(f1wr, f1wi, wf1s, nff);
    pack2_bf16_kernel<<<bff, 256, 0, stream>>>(f2wr, f2wi, wf2s, nff);
  }

  // --- rotary ---
  rotary_kernel<<<Mrows, 256, 0, stream>>>(x_real, x_imag, xrf, xstack);

  const int combBlocks = (int)(((size_t)Mrows * Dc + 255) / 256);

  // --- q/k/v projections: one stacked GEMM each + pack ---
  {
    dim3 g(2 * Dc / BLK_N, 2 * Mrows / BLK_M, 1);
    gemm_bf16_kernel<false><<<g, 256, 0, stream>>>(xstack, wqs, gemmC, 2 * Dc, Dc, 0, 0, 0);
    combine_pack_qk_kernel<<<combBlocks, 256, 0, stream>>>(gemmC, qbr, qbi, qpack, 0);
    gemm_bf16_kernel<false><<<g, 256, 0, stream>>>(xstack, wks, gemmC, 2 * Dc, Dc, 0, 0, 0);
    combine_pack_qk_kernel<<<combBlocks, 256, 0, stream>>>(gemmC, kbr, kbi, kpack, 0);
    gemm_bf16_kernel<false><<<g, 256, 0, stream>>>(xstack, wvs, gemmC, 2 * Dc, Dc, 0, 0, 0);
    combine_pack_qk_kernel<<<combBlocks, 256, 0, stream>>>(gemmC, vbr, vbi, vpackT, 1);
  }

  // --- attention, 4 chunks of 8 (b,h) batches ---
  for (int c = 0; c < BH / CH; ++c) {
    const __bf16* qc = qpack + (size_t)c * CH * Tc * 128;
    const __bf16* kc = kpack + (size_t)c * CH * Tc * 128;
    const __bf16* vc = vpackT + (size_t)c * CH * 128 * Tc;
    float* ac = apack + (size_t)c * CH * Tc * 128;
    // scores = [qr|qi] x [kr|ki]^T  (K=128), bf16 out
    gemm_bf16_kernel<true><<<dim3(Tc / BLK_N, Tc / BLK_M, CH), 256, 0, stream>>>(
        qc, kc, scoresb, Tc, 128,
        (long long)Tc * 128, (long long)Tc * 128, (long long)Tc * Tc);
    softmax_bf16_kernel<<<CH * Tc, 256, 0, stream>>>(scoresb, probsb, 0.125f);
    // attn out = probs x v (v stored transposed 128 x T)
    gemm_bf16_kernel<false><<<dim3(1, Tc / BLK_M, CH), 256, 0, stream>>>(
        probsb, vc, ac, 128, Tc,
        (long long)Tc * Tc, (long long)128 * Tc, (long long)Tc * 128);
  }
  unpack_attn_kernel<<<combBlocks, 256, 0, stream>>>(apack, arstack);

  // --- output projection ---
  {
    dim3 g(2 * Dc / BLK_N, 2 * Mrows / BLK_M, 1);
    gemm_bf16_kernel<false><<<g, 256, 0, stream>>>(arstack, wos, gemmC, 2 * Dc, Dc, 0, 0, 0);
    combine_linear_kernel<<<combBlocks, 256, 0, stream>>>(gemmC, obr, obi, ar2f, ai2f, arstack2, Dc);
  }

  // --- res1 = LN(xr + ar) ---
  add_ln_kernel<<<Mrows, 256, 0, stream>>>(xrf, ar2f, n1g, n1b, res1);

  // --- FFN ---
  {
    dim3 g1(2 * FFc / BLK_N, 2 * Mrows / BLK_M, 1);
    gemm_bf16_kernel<false><<<g1, 256, 0, stream>>>(arstack2, wf1s, gemmC, 2 * FFc, Dc, 0, 0, 0);
    const int gelBlocks = (int)(((size_t)Mrows * FFc + 255) / 256);
    combine_ffn_gelu_kernel<<<gelBlocks, 256, 0, stream>>>(gemmC, f1br, f1bi, mstack);
    dim3 g2(2 * Dc / BLK_N, 2 * Mrows / BLK_M, 1);
    gemm_bf16_kernel<false><<<g2, 256, 0, stream>>>(mstack, wf2s, gemmC, 2 * Dc, FFc, 0, 0, 0);
    combine_linear_kernel<<<combBlocks, 256, 0, stream>>>(gemmC, f2br, f2bi, frf, fif, nullptr, Dc);
  }

  // --- final: res2 = LN(res1 + fr), imag = ai + fi ---
  final_kernel<<<Mrows, 256, 0, stream>>>(res1, frf, ai2f, fif, n2g, n2b, (float*)d_out);
}
